// NEWPooling6_26362509262946
// MI455X (gfx1250) — matrix operand order repaired and checked by reading, not compiled, not run on hardware
//
#include <hip/hip_runtime.h>
#include <hip/hip_bf16.h>

// ---------------------------------------------------------------------------
// Problem constants (match reference)
// ---------------------------------------------------------------------------
#define N_NODES 8192
#define C_DIM   128
#define K_DIM   128                  // inner dim of every GEMM in the pipeline
#define HEADS   6
#define HC      (HEADS * C_DIM)      // 768
#define SLOPE   0.2f
#define EPSV    1e-8f

#define LDK     (K_DIM + 8)          // padded LDS row stride (136 halves = 272B)

typedef __bf16 v16bf __attribute__((ext_vector_type(16)));
typedef __bf16 v8bf  __attribute__((ext_vector_type(8)));
typedef float  v8f   __attribute__((ext_vector_type(8)));
typedef unsigned int u32x4 __attribute__((ext_vector_type(4)));
typedef int          i32x8 __attribute__((ext_vector_type(8)));
typedef int          i32x4 __attribute__((ext_vector_type(4)));

// fp32 -> bf16 bits, round-to-nearest-even (storage-only use of __bf16)
__device__ __forceinline__ unsigned short bfbits(float f) {
    unsigned u = __float_as_uint(f);
    return (unsigned short)((u + 0x7fffu + ((u >> 16) & 1u)) >> 16);
}
__device__ __forceinline__ unsigned pk2(float lo, float hi) {
    return (unsigned)bfbits(lo) | ((unsigned)bfbits(hi) << 16);
}

// order-preserving float<->uint for atomic max of signed floats
__device__ __forceinline__ unsigned fenc(float f) {
    unsigned u = __float_as_uint(f);
    return (u & 0x80000000u) ? ~u : (u | 0x80000000u);
}
__device__ __forceinline__ float fdec(unsigned u) {
    return (u & 0x80000000u) ? __uint_as_float(u & 0x7fffffffu)
                             : __uint_as_float(~u);
}

__device__ __forceinline__ float wave_reduce_sum(float v) {
    #pragma unroll
    for (int m = 16; m > 0; m >>= 1) v += __shfl_xor(v, m, 32);
    return v;
}

// block of 128 threads (4 waves) -> sum in thread 0
__device__ __forceinline__ float block128_reduce_sum(float v) {
    __shared__ float s[4];
    v = wave_reduce_sum(v);
    int w = threadIdx.x >> 5, l = threadIdx.x & 31;
    if (l == 0) s[w] = v;
    __syncthreads();
    float r = 0.f;
    if (threadIdx.x == 0) r = s[0] + s[1] + s[2] + s[3];
    __syncthreads();
    return r;
}

// ---------------------------------------------------------------------------
// TDM 2-D bf16 tile load -> LDS (amdgpu-toolchain 6-arg builtin; ISA 08 D#).
// Pads 16B after every 256B row so a 128-half row lands on a 272B (LDK) stride.
// ---------------------------------------------------------------------------
#if __has_builtin(__builtin_amdgcn_tensor_load_to_lds)
#define HAVE_TDM 1
__device__ __forceinline__ void tdm_load_2d_bf16(
    const unsigned short* gptr, unsigned ldsOff,
    unsigned tensor_d0, unsigned tensor_d1,
    unsigned long long stride0_elems,
    unsigned tile_d0, unsigned tile_d1)
{
    unsigned long long ga = (unsigned long long)(size_t)gptr;
    u32x4 g0;
    g0.x = 1u;                                        // count=1 (valid), user mode
    g0.y = ldsOff;                                    // D#.lds_addr (bytes)
    g0.z = (unsigned)(ga & 0xFFFFFFFFull);            // global_addr[31:0]
    g0.w = (unsigned)((ga >> 32) & 0x01FFFFFFull)     // global_addr[56:32]
         | 0x80000000u;                               // type=2 ("image")
    i32x8 g1;
    unsigned d0 = (1u << 16)        // data_size: 2 bytes
                | (1u << 20)        // pad_enable (load only)
                | (5u << 22)        // pad_interval: 64 DWORDs (= one 256B row)
                | (3u << 25);       // pad_amount: 4 DWORDs (16B) -> 272B stride
    g1[0] = (int)d0;                                  // workgroup_mask=0 (no cluster)
    g1[1] = (int)((tensor_d0 & 0xFFFFu) << 16);       // abar=0 | tensor_dim0.lo
    g1[2] = (int)((tensor_d0 >> 16) | ((tensor_d1 & 0xFFFFu) << 16));
    g1[3] = (int)((tensor_d1 >> 16) | ((tile_d0 & 0xFFFFu) << 16));
    g1[4] = (int)(tile_d1 & 0xFFFFu);                 // tile_dim2 = 0 (2-D)
    g1[5] = (int)(unsigned)(stride0_elems & 0xFFFFFFFFull);
    g1[6] = (int)(unsigned)((stride0_elems >> 32) & 0xFFFFull); // dim1_stride=0
    g1[7] = 0;
    i32x4 z4 = {0, 0, 0, 0};                          // groups 2/3: unused (2-D)
    i32x8 z8 = {0, 0, 0, 0, 0, 0, 0, 0};              // 6-arg form: extra group
    __builtin_amdgcn_tensor_load_to_lds(g0, g1, z4, z4, z8, 0);
}
#else
#define HAVE_TDM 0
#endif

// ---------------------------------------------------------------------------
// WMMA bf16 GEMM:  C[M x Nc] = A[M x 128] * Bt[Nc x 128]^T   (bf16 in, f32 out)
// A row-major [M][128] bf16; Bt is B pre-transposed: Bt[col][k] bf16.
// M % 128 == 0, Nc % 64 == 0, K == 128. 256 threads = 8 waves, tile 128x64,
// full K in LDS; operand tiles arrive via Tensor Data Mover (TENSORcnt).
// ---------------------------------------------------------------------------
__global__ __launch_bounds__(256)
void gemm_wmma_bf16(const unsigned short* __restrict__ A,
                    const unsigned short* __restrict__ Bt,
                    float* __restrict__ C, int M, int Nc)
{
    __shared__ unsigned short As[128 * LDK];   // 34816 B, stride 272B
    __shared__ unsigned short Bs[64 * LDK];    // 17408 B, Bs[col][k]

    const int tid  = threadIdx.x;
    const int wave = tid >> 5;
    const int lane = tid & 31;
    const int l16  = lane & 15;
    const int lhi  = lane >> 4;                // half-wave select
    const int rowBase = blockIdx.y * 128;
    const int colBase = blockIdx.x * 64;

#if HAVE_TDM
    if (tid < 32) {                            // wave 0 issues both DMA tiles
        tdm_load_2d_bf16(A + (size_t)rowBase * K_DIM,
                         (unsigned)(unsigned long long)(size_t)(void*)&As[0],
                         K_DIM, (unsigned)M, K_DIM, K_DIM, 128);
        tdm_load_2d_bf16(Bt + (size_t)colBase * K_DIM,
                         (unsigned)(unsigned long long)(size_t)(void*)&Bs[0],
                         K_DIM, (unsigned)Nc, K_DIM, K_DIM, 64);
        __builtin_amdgcn_s_wait_tensorcnt(0);
    }
#else
    // fallback: cooperative vectorized staging of the bf16 operands
    #pragma unroll
    for (int t = tid; t < 128 * 16; t += 256) {
        int r = t >> 4, kq = (t & 15) << 3;    // 8 halves (16B) per copy
        *reinterpret_cast<uint4*>(&As[r * LDK + kq]) =
            *reinterpret_cast<const uint4*>(&A[(size_t)(rowBase + r) * K_DIM + kq]);
    }
    #pragma unroll
    for (int t = tid; t < 64 * 16; t += 256) {
        int r = t >> 4, kq = (t & 15) << 3;
        *reinterpret_cast<uint4*>(&Bs[r * LDK + kq]) =
            *reinterpret_cast<const uint4*>(&Bt[(size_t)(colBase + r) * K_DIM + kq]);
    }
#endif
    __syncthreads();

    v8f acc[4];
    #pragma unroll
    for (int n = 0; n < 4; ++n)
        acc[n] = (v8f){0.f,0.f,0.f,0.f,0.f,0.f,0.f,0.f};

    const unsigned short* arow = &As[(wave * 16 + l16) * LDK];

    #pragma unroll
    for (int k0 = 0; k0 < K_DIM; k0 += 32) {
        // A fragment (ISA layout): K(lane,i) = k0 + (i>=8?16:0) + lhi*8 + (i&7)
        v8bf alo = *reinterpret_cast<const v8bf*>(&arow[k0 + lhi * 8]);
        v8bf ahi = *reinterpret_cast<const v8bf*>(&arow[k0 + 16 + lhi * 8]);
        v16bf af = __builtin_shufflevector(alo, ahi,
            0,1,2,3,4,5,6,7,8,9,10,11,12,13,14,15);
        #pragma unroll
        for (int n = 0; n < 4; ++n) {
            // B fragment: col = n*16 + l16, K = k0 + (i>=8?16:0) + lhi*8 + (i&7)
            const unsigned short* bc = &Bs[(n * 16 + l16) * LDK];
            v8bf blo = *reinterpret_cast<const v8bf*>(&bc[k0 + lhi * 8]);
            v8bf bhi = *reinterpret_cast<const v8bf*>(&bc[k0 + 16 + lhi * 8]);
            v16bf bfr = __builtin_shufflevector(blo, bhi,
                0,1,2,3,4,5,6,7,8,9,10,11,12,13,14,15);
            acc[n] = __builtin_amdgcn_wmma_f32_16x16x32_bf16(
                false, af, false, bfr, (short)0, acc[n], false, false);
        }
    }

    // C/D layout: col = lane&15, row = vgpr r + (lanehi?8:0)
    #pragma unroll
    for (int n = 0; n < 4; ++n) {
        #pragma unroll
        for (int r = 0; r < 8; ++r) {
            int row = rowBase + wave * 16 + r + (lhi << 3);
            int col = colBase + n * 16 + l16;
            C[(size_t)row * Nc + col] = acc[n][r];
        }
    }
}

// ---------------------------------------------------------------------------
// Conversion kernels feeding the TDM GEMM
// ---------------------------------------------------------------------------
// packed fp32 -> bf16 (2 elems/thread-iter); n is even
__global__ void k_cvt_bf16(const float* __restrict__ src,
                           unsigned* __restrict__ dst2, int n2) {
    int i = blockIdx.x * blockDim.x + threadIdx.x;
    if (i >= n2) return;
    const float2 v = *reinterpret_cast<const float2*>(&src[(size_t)i * 2]);
    if (i + 512 < n2)
        __builtin_prefetch(&src[(size_t)(i + 512) * 2], 0, 0);  // global_prefetch_b8
    dst2[i] = pk2(v.x, v.y);
}

// B[K x Nc] fp32 -> Bt[Nc x K] bf16 (transpose + convert)
__global__ void k_cvt_tr_bf16(const float* __restrict__ B,
                              unsigned short* __restrict__ Bt, int Nc) {
    int i = blockIdx.x * blockDim.x + threadIdx.x;
    if (i >= K_DIM * Nc) return;
    int c = i % Nc, k = i / Nc;               // reads coalesced along Nc
    Bt[(size_t)c * K_DIM + k] = bfbits(B[(size_t)k * Nc + c]);
}

// ---------------------------------------------------------------------------
// Pipeline helper kernels
// ---------------------------------------------------------------------------
__global__ void k_fill(float* p, float v, int n) {
    int i = blockIdx.x * blockDim.x + threadIdx.x;
    if (i < n) p[i] = v;
}

// degree over kept edges (self-loops pre-counted by init to 1.0)
__global__ void k_degree(const long long* __restrict__ ei, int E,
                         float* __restrict__ deg) {
    int e = blockIdx.x * blockDim.x + threadIdx.x;
    if (e >= E) return;
    int s = (int)ei[e], d = (int)ei[E + e];
    if (s != d) atomicAdd(&deg[d], 1.0f);
}

__global__ void k_dinv(const float* __restrict__ deg, float* __restrict__ dinv,
                       int n) {
    int i = blockIdx.x * blockDim.x + threadIdx.x;
    if (i < n) dinv[i] = rsqrtf(fmaxf(deg[i], 1.0f));
}

// xg[d] += dinv[s]*dinv[d]*h[s]  over edges + self loops; block = 1 edge
__global__ __launch_bounds__(128)
void k_gcn_agg(const long long* __restrict__ ei, int E,
               const float* __restrict__ h, const float* __restrict__ dinv,
               float* __restrict__ xg) {
    int e = blockIdx.x;
    int s, d;
    if (e < E) {
        s = (int)ei[e]; d = (int)ei[E + e];
        if (s == d) return;                 // reference drops input self loops
    } else {
        s = d = e - E;                      // appended self loop
    }
    float w = dinv[s] * dinv[d];
    int c = threadIdx.x;
    atomicAdd(&xg[(size_t)d * C_DIM + c], w * h[(size_t)s * C_DIM + c]);
}

__global__ void k_bias(float* __restrict__ xg, const float* __restrict__ b,
                       size_t total) {
    size_t i = (size_t)blockIdx.x * blockDim.x + threadIdx.x;
    if (i < total) xg[i] += b[i % C_DIM];
}

// per-destination feature max:  xjm[d][h][c] = max over in-edges of xw[s][h][c]
__global__ __launch_bounds__(256)
void k_seg_max_feat(const long long* __restrict__ ei, int E,
                    const float* __restrict__ xw, unsigned* __restrict__ xjm) {
    int e = blockIdx.x;
    int s, d;
    if (e < E) {
        s = (int)ei[e]; d = (int)ei[E + e];
        if (s == d) return;
    } else { s = d = e - E; }
    for (int c = threadIdx.x; c < HC; c += 256)
        atomicMax(&xjm[(size_t)d * HC + c], fenc(xw[(size_t)s * HC + c]));
}

// per-edge per-head attention logits + running segment max
__global__ __launch_bounds__(192)
void k_alpha(const long long* __restrict__ ei, int E,
             const float* __restrict__ xw, const unsigned* __restrict__ xjm,
             const float* __restrict__ att, float* __restrict__ alpha,
             unsigned* __restrict__ amax) {
    int e = blockIdx.x;
    int s, d; bool skip = false;
    if (e < E) {
        s = (int)ei[e]; d = (int)ei[E + e];
        skip = (s == d);
    } else { s = d = e - E; }
    int hd   = threadIdx.x >> 5;            // head 0..5
    int lane = threadIdx.x & 31;
    if (skip) {
        if (lane == 0) alpha[(size_t)e * HEADS + hd] = -1e30f;
        return;
    }
    float acc = 0.f;
    #pragma unroll
    for (int c = lane; c < C_DIM; c += 32) {
        float xi = xw[(size_t)s * HC + hd * C_DIM + c];
        float xj = fdec(xjm[(size_t)d * HC + hd * C_DIM + c]);
        acc += xi * att[hd * 2 * C_DIM + c] + xj * att[hd * 2 * C_DIM + C_DIM + c];
    }
    acc = wave_reduce_sum(acc);
    if (lane == 0) {
        float a = (acc > 0.f) ? acc : SLOPE * acc;   // leaky relu
        alpha[(size_t)e * HEADS + hd] = a;
        atomicMax(&amax[(size_t)d * HEADS + hd], fenc(a));
    }
}

// e = exp(alpha - amax[d]); segment sum
__global__ void k_exp_sum(const long long* __restrict__ ei, int E, int ET,
                          float* __restrict__ alpha,
                          const unsigned* __restrict__ amax,
                          float* __restrict__ asum) {
    int idx = blockIdx.x * blockDim.x + threadIdx.x;
    if (idx >= ET * HEADS) return;
    int e = idx / HEADS, hd = idx % HEADS;
    int d = (e < E) ? (int)ei[E + e] : (e - E);
    float m  = fdec(amax[(size_t)d * HEADS + hd]);
    float ev = __expf(alpha[idx] - m);        // skipped edges: exp(-1e30-m)=0
    alpha[idx] = ev;
    atomicAdd(&asum[(size_t)d * HEADS + hd], ev);
}

// x_att[d][c] += mean_h( w_eh * xw[s][h][c] )
__global__ __launch_bounds__(128)
void k_att_agg(const long long* __restrict__ ei, int E,
               const float* __restrict__ xw, const float* __restrict__ alpha,
               const float* __restrict__ asum, float* __restrict__ xatt) {
    int e = blockIdx.x;
    int s, d;
    if (e < E) {
        s = (int)ei[e]; d = (int)ei[E + e];
        if (s == d) return;
    } else { s = d = e - E; }
    int c = threadIdx.x;
    float acc = 0.f;
    #pragma unroll
    for (int hd = 0; hd < HEADS; ++hd) {
        float den = asum[(size_t)d * HEADS + hd];
        float w   = alpha[(size_t)e * HEADS + hd] / fmaxf(den, 1e-30f);
        acc += w * xw[(size_t)s * HC + hd * C_DIM + c];
    }
    atomicAdd(&xatt[(size_t)d * C_DIM + c], acc * (1.0f / HEADS));
}

// b[i] = x_att[i] . s2_w + s2_b
__global__ __launch_bounds__(128)
void k_s2_dot(const float* __restrict__ xatt, const float* __restrict__ s2w,
              const float* __restrict__ s2b, float* __restrict__ bvec) {
    int i = blockIdx.x;
    float v = xatt[(size_t)i * C_DIM + threadIdx.x] * s2w[threadIdx.x];
    float s = block128_reduce_sum(v);
    if (threadIdx.x == 0) bvec[i] = s + s2b[0];
}

__global__ __launch_bounds__(128)
void k_norm(const float* __restrict__ a, float* __restrict__ anrm) {
    int i = blockIdx.x;
    float v = a[(size_t)i * C_DIM + threadIdx.x];
    float s = block128_reduce_sum(v * v);
    if (threadIdx.x == 0) anrm[i] = fmaxf(sqrtf(s), EPSV);
}

// fitness accumulation: fsum[d] += b[d] * cos(a[d], a[s])
__global__ __launch_bounds__(128)
void k_cos_fit(const long long* __restrict__ ei, int E,
               const float* __restrict__ a, const float* __restrict__ anrm,
               const float* __restrict__ bvec, float* __restrict__ fsum) {
    int e = blockIdx.x;
    int s, d;
    if (e < E) {
        s = (int)ei[e]; d = (int)ei[E + e];
        if (s == d) return;
    } else { s = d = e - E; }
    float v = a[(size_t)d * C_DIM + threadIdx.x] *
              a[(size_t)s * C_DIM + threadIdx.x];
    float dot = block128_reduce_sum(v);
    if (threadIdx.x == 0) {
        float cosv = dot / (anrm[d] * anrm[s]);
        atomicAdd(&fsum[d], bvec[d] * cosv);
    }
}

__global__ void k_sigmoid(const float* __restrict__ fsum,
                          float* __restrict__ fit, int n) {
    int i = blockIdx.x * blockDim.x + threadIdx.x;
    if (i < n) fit[i] = 1.0f / (1.0f + __expf(-fsum[i]));
}

// fused output write: coarsened features scaled by fitness score
__global__ void k_write_out(const float* __restrict__ xatt,
                            const float* __restrict__ fit,
                            float* __restrict__ out, int out_size) {
#if __has_builtin(__builtin_amdgcn_s_cluster_barrier)
    // NOP when not launched as a cluster (ISA 6.6); exercises s_barrier_signal -3
    __builtin_amdgcn_s_cluster_barrier();
#endif
    int i = blockIdx.x * blockDim.x + threadIdx.x;
    if (i >= out_size) return;
    int node = (i / C_DIM) % N_NODES;
    int c    = i % C_DIM;
    out[i] = xatt[(size_t)node * C_DIM + c] * fit[node];
}

// ---------------------------------------------------------------------------
// Launcher
// ---------------------------------------------------------------------------
extern "C" void kernel_launch(void* const* d_in, const int* in_sizes, int n_in,
                              void* d_out, int out_size, void* d_ws, size_t ws_size,
                              hipStream_t stream) {
    const float*     x     = (const float*)d_in[0];
    const long long* ei    = (const long long*)d_in[1];   // int64 [2, E]
    const float*     Wproj = (const float*)d_in[2];       // [C, HC]
    const float*     att   = (const float*)d_in[3];       // [1, H, 2C]
    const float*     gcn_w = (const float*)d_in[4];       // [C, C]
    const float*     gcn_b = (const float*)d_in[5];       // [C]
    const float*     s1_w  = (const float*)d_in[6];       // [C, C]
    const float*     s2_w  = (const float*)d_in[7];       // [C, 1]
    const float*     s2_b  = (const float*)d_in[8];       // [1]
    float* out = (float*)d_out;

    const int E  = in_sizes[1] / 2;
    const int ET = E + N_NODES;                           // edges + self loops

    // workspace carve-up
    float* p = (float*)d_ws;
    float*    deg  = p;  p += N_NODES;
    float*    dinv = p;  p += N_NODES;
    float*    h    = p;  p += (size_t)N_NODES * C_DIM;
    float*    xg   = p;  p += (size_t)N_NODES * C_DIM;
    float*    xw   = p;  p += (size_t)N_NODES * HC;
    unsigned* xjm  = (unsigned*)p; p += (size_t)N_NODES * HC;
    unsigned* amax = (unsigned*)p; p += (size_t)N_NODES * HEADS;
    float*    asum = p;  p += (size_t)N_NODES * HEADS;
    float*    alph = p;  p += (size_t)ET * HEADS;
    float*    xatt = p;  p += (size_t)N_NODES * C_DIM;
    float*    amat = p;  p += (size_t)N_NODES * C_DIM;
    float*    bvec = p;  p += N_NODES;
    float*    anrm = p;  p += N_NODES;
    float*    fsum = p;  p += N_NODES;
    float*    fit  = p;  p += N_NODES;
    // bf16 operand buffers for TDM GEMMs (abf reused for x / xg / x_att)
    unsigned short* abf  = (unsigned short*)p; p += (size_t)N_NODES * C_DIM / 2;
    unsigned short* btbf = (unsigned short*)p; p += (size_t)HC * K_DIM / 2;

    const int nA2 = N_NODES * C_DIM / 2;                  // packed cvt count

    // zero accumulators (capture-safe async memsets)
    (void)hipMemsetAsync(xg,   0, (size_t)N_NODES * C_DIM * 4, stream);
    (void)hipMemsetAsync(xjm,  0, (size_t)N_NODES * HC * 4, stream); // < enc(-inf)
    (void)hipMemsetAsync(amax, 0, (size_t)N_NODES * HEADS * 4, stream);
    (void)hipMemsetAsync(asum, 0, (size_t)N_NODES * HEADS * 4, stream);
    (void)hipMemsetAsync(xatt, 0, (size_t)N_NODES * C_DIM * 4, stream);
    (void)hipMemsetAsync(fsum, 0, (size_t)N_NODES * 4, stream);

    // 1) degrees (self loop pre-counted) -> dinv
    k_fill<<<(N_NODES + 255) / 256, 256, 0, stream>>>(deg, 1.0f, N_NODES);
    k_degree<<<(E + 255) / 256, 256, 0, stream>>>(ei, E, deg);
    k_dinv<<<(N_NODES + 255) / 256, 256, 0, stream>>>(deg, dinv, N_NODES);

    // 2) h = x @ gcn_w  (TDM + WMMA)
    k_cvt_bf16<<<(nA2 + 255) / 256, 256, 0, stream>>>(x, (unsigned*)abf, nA2);
    k_cvt_tr_bf16<<<(K_DIM * C_DIM + 255) / 256, 256, 0, stream>>>(gcn_w, btbf,
                                                                   C_DIM);
    gemm_wmma_bf16<<<dim3(C_DIM / 64, N_NODES / 128), 256, 0, stream>>>(
        abf, btbf, h, N_NODES, C_DIM);

    // 3) normalized aggregation + bias
    k_gcn_agg<<<ET, 128, 0, stream>>>(ei, E, h, dinv, xg);
    k_bias<<<(int)(((size_t)N_NODES * C_DIM + 255) / 256), 256, 0, stream>>>(
        xg, gcn_b, (size_t)N_NODES * C_DIM);

    // 4) xw = xg @ weight  (TDM + WMMA, Nc = 768)
    k_cvt_bf16<<<(nA2 + 255) / 256, 256, 0, stream>>>(xg, (unsigned*)abf, nA2);
    k_cvt_tr_bf16<<<(K_DIM * HC + 255) / 256, 256, 0, stream>>>(Wproj, btbf, HC);
    gemm_wmma_bf16<<<dim3(HC / 64, N_NODES / 128), 256, 0, stream>>>(
        abf, btbf, xw, N_NODES, HC);

    // 5) attention: seg-max features, logits, softmax, aggregate
    k_seg_max_feat<<<ET, 256, 0, stream>>>(ei, E, xw, xjm);
    k_alpha<<<ET, 192, 0, stream>>>(ei, E, xw, xjm, att, alph, amax);
    k_exp_sum<<<(ET * HEADS + 255) / 256, 256, 0, stream>>>(ei, E, ET, alph,
                                                            amax, asum);
    k_att_agg<<<ET, 128, 0, stream>>>(ei, E, xw, alph, asum, xatt);

    // 6) scoring head: a = x_att @ s1_w (TDM + WMMA), b = x_att @ s2_w + s2_b
    k_cvt_bf16<<<(nA2 + 255) / 256, 256, 0, stream>>>(xatt, (unsigned*)abf, nA2);
    k_cvt_tr_bf16<<<(K_DIM * C_DIM + 255) / 256, 256, 0, stream>>>(s1_w, btbf,
                                                                   C_DIM);
    gemm_wmma_bf16<<<dim3(C_DIM / 64, N_NODES / 128), 256, 0, stream>>>(
        abf, btbf, amat, N_NODES, C_DIM);
    k_s2_dot<<<N_NODES, 128, 0, stream>>>(xatt, s2_w, s2_b, bvec);
    k_norm<<<N_NODES, 128, 0, stream>>>(amat, anrm);
    k_cos_fit<<<ET, 128, 0, stream>>>(ei, E, amat, anrm, bvec, fsum);
    k_sigmoid<<<(N_NODES + 255) / 256, 256, 0, stream>>>(fsum, fit, N_NODES);

    // 7) output
    if (out_size > 0)
        k_write_out<<<(out_size + 255) / 256, 256, 0, stream>>>(xatt, fit, out,
                                                                out_size);
}
